// MQAttention_72988674228288
// MI455X (gfx1250) — matrix operand order repaired and checked by reading, not compiled, not run on hardware
//
#include <hip/hip_runtime.h>
#include <hip/hip_bf16.h>

#define EMBED  1024
#define SLEN   4096
#define NHEADS 16
#define HD     64

typedef __attribute__((ext_vector_type(16))) __bf16 v16bf;
typedef __attribute__((ext_vector_type(8)))  float  v8f;

union Frag16 {
    v16bf v;
    uint4 q[2];
    unsigned short u[16];
};

__device__ __forceinline__ unsigned short f2bf(float f) {
    unsigned int u = __float_as_uint(f);
    u += 0x7FFFu + ((u >> 16) & 1u);          // round-to-nearest-even
    return (unsigned short)(u >> 16);
}

__device__ __forceinline__ v8f wmma_bf16(const Frag16& a, const Frag16& b, v8f c) {
    return __builtin_amdgcn_wmma_f32_16x16x32_bf16(
        /*neg_a=*/false, a.v, /*neg_b=*/false, b.v,
        /*c_mod=*/(short)0, c, /*reuse_a=*/false, /*reuse_b=*/false);
}

__global__ void cvt_f32_to_bf16(const float* __restrict__ in,
                                unsigned short* __restrict__ out, int n) {
    int i = blockIdx.x * blockDim.x + threadIdx.x;
    if (i < n) out[i] = f2bf(in[i]);
}

// in[K][N] f32 -> out[N][K] bf16 (weights are small; done once)
__global__ void cvt_transpose_bf16(const float* __restrict__ in,
                                   unsigned short* __restrict__ out,
                                   int K, int N) {
    int i = blockIdx.x * blockDim.x + threadIdx.x;
    if (i < K * N) {
        int k = i / N, n = i % N;
        out[(size_t)n * K + k] = f2bf(in[i]);
    }
}

// C[M,N] = A[M,K]bf16(row-major) x Bt[N,K]bf16 (pre-transposed) + bias[N]
// Each wave computes a 16x64 strip; register double-buffering hides load latency.
__global__ void __launch_bounds__(256)
gemm_bf16(const unsigned short* __restrict__ A,
          const unsigned short* __restrict__ Bt,
          const float* __restrict__ bias,
          unsigned short* __restrict__ outBF,
          float* __restrict__ outF,
          int M, int N, int K)
{
    const int lane = threadIdx.x & 31;
    const int wave = threadIdx.x >> 5;
    const int strip = blockIdx.x * 8 + wave;
    const int nstrip = N >> 6;
    const int tm = (strip / nstrip) << 4;
    const int tn = (strip % nstrip) << 6;
    const int nn   = lane & 15;
    const int half = (lane >> 4) & 1;
    const int hi8  = half << 3;

    const unsigned short* arow = A + (size_t)(tm + nn) * K;
    const unsigned short* brow[4];
#pragma unroll
    for (int t = 0; t < 4; ++t)
        brow[t] = Bt + (size_t)(tn + (t << 4) + nn) * K + hi8;

    // prologue: fragments for kk = 0
    Frag16 a, b[4];
    a.q[0] = *(const uint4*)(arow + hi8);
    a.q[1] = *(const uint4*)(arow + hi8 + 16);
#pragma unroll
    for (int t = 0; t < 4; ++t) {
        b[t].q[0] = *(const uint4*)(brow[t]);
        b[t].q[1] = *(const uint4*)(brow[t] + 16);
    }

    v8f c[4] = {};
    for (int kk = 0; kk < K; kk += 32) {
        // prefetch next K-step (clamped wrap keeps addresses valid)
        const int kn = (kk + 32 < K) ? kk + 32 : 0;
        Frag16 an, bn[4];
        an.q[0] = *(const uint4*)(arow + kn + hi8);
        an.q[1] = *(const uint4*)(arow + kn + hi8 + 16);
#pragma unroll
        for (int t = 0; t < 4; ++t) {
            bn[t].q[0] = *(const uint4*)(brow[t] + kn);
            bn[t].q[1] = *(const uint4*)(brow[t] + kn + 16);
        }
        // compute on current fragments (loads above are still in flight)
#pragma unroll
        for (int t = 0; t < 4; ++t)
            c[t] = wmma_bf16(a, b[t], c[t]);
        a = an;
#pragma unroll
        for (int t = 0; t < 4; ++t) b[t] = bn[t];
    }

#pragma unroll
    for (int t = 0; t < 4; ++t) {
        const float bb = bias[tn + (t << 4) + nn];
#pragma unroll
        for (int r = 0; r < 8; ++r) {
            float vv = c[t][r] + bb;
            size_t idx = (size_t)(tm + r + hi8) * N + tn + (t << 4) + nn;
            if (outBF) outBF[idx] = f2bf(vv);
            if (outF)  outF[idx]  = vv;
        }
    }
}

// All flash-kernel LDS in one object -> well-defined byte offsets from LDS base 0,
// as required by the async-to-LDS VDST operand. K/V are double-buffered.
struct FlashSmem {
    unsigned short k[4][2][32][64];  // [wave][buf][key][d]
    unsigned short v[4][2][32][64];  // [wave][buf][key][d]
    unsigned short p[4][16][32];     // P tile relay (C-layout -> A-layout)
};
#define KV_BUF_BYTES (32 * 64 * 2)

// Flash MQA: one 16-row Q block per wave, 4 waves/block, grid (SLEN/64, NHEADS).
__global__ void __launch_bounds__(128)
mqa_flash(const unsigned short* __restrict__ Q,   // [SLEN][EMBED], head h at cols h*64
          const unsigned short* __restrict__ Km,  // [SLEN][HD]
          const unsigned short* __restrict__ Vm,  // [SLEN][HD]
          const int* __restrict__ mask,           // [SLEN][SLEN]
          unsigned short* __restrict__ O)         // [SLEN][EMBED] bf16
{
    __shared__ FlashSmem smem;

    const int lane = threadIdx.x & 31;
    const int w    = threadIdx.x >> 5;
    const int head = blockIdx.y;
    const int q0   = (blockIdx.x * 4 + w) << 4;
    const int nn   = lane & 15;
    const int half = (lane >> 4) & 1;
    const int hi8  = half << 3;

    // Q A-fragments for d=0..31 and d=32..63
    Frag16 aq0, aq1;
    {
        const unsigned short* qrow = Q + (size_t)(q0 + nn) * EMBED + head * HD;
        aq0.q[0] = *(const uint4*)(qrow + (half << 4));
        aq0.q[1] = *(const uint4*)(qrow + (half << 4) + 8);
        aq1.q[0] = *(const uint4*)(qrow + 32 + (half << 4));
        aq1.q[1] = *(const uint4*)(qrow + 32 + (half << 4) + 8);
    }

    const unsigned kbase = (unsigned)((char*)&smem.k[w][0][lane][0] - (char*)&smem);
    const unsigned vbase = (unsigned)((char*)&smem.v[w][0][lane][0] - (char*)&smem);

    // async DMA of one 32-key K/V chunk into LDS buffer `buf`
    auto issue_stage = [&](int kvChunk, int buf) {
        const unsigned ko = kbase + buf * KV_BUF_BYTES;
        const unsigned vo = vbase + buf * KV_BUF_BYTES;
        const unsigned long long ka =
            (unsigned long long)(Km + (size_t)(kvChunk + lane) * HD);
        const unsigned long long va =
            (unsigned long long)(Vm + (size_t)(kvChunk + lane) * HD);
        asm volatile(
            "global_load_async_to_lds_b128 %0, %2, off\n\t"
            "global_load_async_to_lds_b128 %0, %2, off offset:16\n\t"
            "global_load_async_to_lds_b128 %0, %2, off offset:32\n\t"
            "global_load_async_to_lds_b128 %0, %2, off offset:48\n\t"
            "global_load_async_to_lds_b128 %1, %3, off\n\t"
            "global_load_async_to_lds_b128 %1, %3, off offset:16\n\t"
            "global_load_async_to_lds_b128 %1, %3, off offset:32\n\t"
            "global_load_async_to_lds_b128 %1, %3, off offset:48"
            :: "v"(ko), "v"(vo), "v"(ka), "v"(va)
            : "memory");
    };

    v8f o[4] = {};
    float mrow[8], lrow[8];
#pragma unroll
    for (int r = 0; r < 8; ++r) { mrow[r] = -1e30f; lrow[r] = 0.0f; }

    issue_stage(0, 0);                      // prologue DMA

    for (int i = 0; i < SLEN / 32; ++i) {
        const int kv  = i << 5;
        const int cur = i & 1;

        // prefetch next chunk into the other buffer (uniform wraparound keeps
        // the pipeline shape constant; last prefetch is harmless & unused)
        {
            int nxt = kv + 32;
            if (nxt >= SLEN) nxt = 0;
            asm volatile("s_wait_dscnt 0" ::: "memory");   // old reads drained
            issue_stage(nxt, cur ^ 1);
        }
        // async loads complete in order: <=8 outstanding => chunk i has landed
        asm volatile("s_wait_asynccnt 8" ::: "memory");

        // ---- S = Q K^T for two 16-key tiles ----
        v8f sc[2];
#pragma unroll
        for (int t = 0; t < 2; ++t) {
            const unsigned short* kr = &smem.k[w][cur][(t << 4) + nn][0];
            v8f s = {};
            Frag16 b;
            b.q[0] = *(const uint4*)(kr + (half << 4));       // d = 16*half + 0..7
            b.q[1] = *(const uint4*)(kr + (half << 4) + 8);   // d = 16*half + 8..15
            s = wmma_bf16(aq0, b, s);
            b.q[0] = *(const uint4*)(kr + 32 + (half << 4));
            b.q[1] = *(const uint4*)(kr + 32 + (half << 4) + 8);
            s = wmma_bf16(aq1, b, s);
            sc[t] = s;
        }

        // ---- scale, mask, online softmax update ----
#pragma unroll
        for (int r = 0; r < 8; ++r) {
            const int gq = q0 + r + hi8;
            float v0 = sc[0][r] * 0.125f;   // 1/sqrt(64)
            float v1 = sc[1][r] * 0.125f;
            if (mask[(size_t)gq * SLEN + kv + nn] == 0)      v0 = -10000.0f;
            if (mask[(size_t)gq * SLEN + kv + 16 + nn] == 0) v1 = -10000.0f;
            float rm = fmaxf(v0, v1);
            rm = fmaxf(rm, __shfl_xor(rm, 1, 32));
            rm = fmaxf(rm, __shfl_xor(rm, 2, 32));
            rm = fmaxf(rm, __shfl_xor(rm, 4, 32));
            rm = fmaxf(rm, __shfl_xor(rm, 8, 32));
            const float mnew  = fmaxf(mrow[r], rm);
            const float alpha = __expf(mrow[r] - mnew);
            const float p0 = __expf(v0 - mnew);
            const float p1 = __expf(v1 - mnew);
            float rs = p0 + p1;
            rs += __shfl_xor(rs, 1, 32);
            rs += __shfl_xor(rs, 2, 32);
            rs += __shfl_xor(rs, 4, 32);
            rs += __shfl_xor(rs, 8, 32);
            lrow[r] = lrow[r] * alpha + rs;
            mrow[r] = mnew;
            smem.p[w][r + hi8][nn]      = f2bf(p0);
            smem.p[w][r + hi8][16 + nn] = f2bf(p1);
            o[0][r] *= alpha; o[1][r] *= alpha; o[2][r] *= alpha; o[3][r] *= alpha;
        }
        asm volatile("s_wait_dscnt 0" ::: "memory");  // cross-lane P relay

        // ---- O += P V ----
        Frag16 ap;
        {
            const unsigned short* pr = &smem.p[w][nn][0];
            ap.q[0] = *(const uint4*)(pr + (half << 4));
            ap.q[1] = *(const uint4*)(pr + (half << 4) + 8);
        }
#pragma unroll
        for (int t = 0; t < 4; ++t) {
            Frag16 bv;  // B element (k=key 16*half+j, n=d t*16+nn)
#pragma unroll
            for (int j = 0; j < 16; ++j)
                bv.u[j] = smem.v[w][cur][(half << 4) + j][(t << 4) + nn];
            o[t] = wmma_bf16(ap, bv, o[t]);
        }
    }

    // ---- normalize and store (bf16, feeds output projection) ----
#pragma unroll
    for (int r = 0; r < 8; ++r) {
        const float inv = 1.0f / lrow[r];
        unsigned short* dst = O + (size_t)(q0 + r + hi8) * EMBED + head * HD;
        dst[nn]      = f2bf(o[0][r] * inv);
        dst[16 + nn] = f2bf(o[1][r] * inv);
        dst[32 + nn] = f2bf(o[2][r] * inv);
        dst[48 + nn] = f2bf(o[3][r] * inv);
    }
}

extern "C" void kernel_launch(void* const* d_in, const int* in_sizes, int n_in,
                              void* d_out, int out_size, void* d_ws, size_t ws_size,
                              hipStream_t stream)
{
    (void)in_sizes; (void)n_in; (void)out_size; (void)ws_size;
    const float* x    = (const float*)d_in[0];
    const int*   mask = (const int*)  d_in[1];
    const float* Wq   = (const float*)d_in[2];
    const float* bq   = (const float*)d_in[3];
    const float* Wk   = (const float*)d_in[4];
    const float* bk   = (const float*)d_in[5];
    const float* Wv   = (const float*)d_in[6];
    const float* bv   = (const float*)d_in[7];
    const float* Wo   = (const float*)d_in[8];
    const float* bo   = (const float*)d_in[9];
    float* out = (float*)d_out;

    char* ws = (char*)d_ws;
    size_t off = 0;
    auto bump = [&](size_t elems) {
        unsigned short* p = (unsigned short*)(ws + off);
        off += (elems * sizeof(unsigned short) + 255) & ~(size_t)255;
        return p;
    };
    unsigned short* xh   = bump((size_t)SLEN * EMBED);   // x bf16 [S][E]
    unsigned short* Wqt  = bump((size_t)EMBED * EMBED);  // Wq^T bf16 [E][E]
    unsigned short* Wkt  = bump((size_t)HD * EMBED);     // Wk^T bf16 [64][E]
    unsigned short* Wvt  = bump((size_t)HD * EMBED);     // Wv^T bf16 [64][E]
    unsigned short* Wot  = bump((size_t)EMBED * EMBED);  // Wo^T bf16 [E][E]
    unsigned short* Qh   = bump((size_t)SLEN * EMBED);   // Q bf16
    unsigned short* Kh   = bump((size_t)SLEN * HD);      // K bf16
    unsigned short* Vh   = bump((size_t)SLEN * HD);      // V bf16
    unsigned short* Ah   = bump((size_t)SLEN * EMBED);   // attn out bf16

    cvt_f32_to_bf16<<<(SLEN * EMBED + 255) / 256, 256, 0, stream>>>(x, xh, SLEN * EMBED);
    cvt_transpose_bf16<<<(EMBED * EMBED + 255) / 256, 256, 0, stream>>>(Wq, Wqt, EMBED, EMBED);
    cvt_transpose_bf16<<<(EMBED * HD + 255) / 256, 256, 0, stream>>>(Wk, Wkt, EMBED, HD);
    cvt_transpose_bf16<<<(EMBED * HD + 255) / 256, 256, 0, stream>>>(Wv, Wvt, EMBED, HD);
    cvt_transpose_bf16<<<(EMBED * EMBED + 255) / 256, 256, 0, stream>>>(Wo, Wot, EMBED, EMBED);

    // Projections (bf16 outputs, f32 accumulate). Waves own 16x64 strips.
    gemm_bf16<<<(SLEN / 16) * (EMBED / 64) / 8, 256, 0, stream>>>(
        xh, Wqt, bq, Qh, (float*)nullptr, SLEN, EMBED, EMBED);
    gemm_bf16<<<(SLEN / 16) * (HD / 64) / 8, 256, 0, stream>>>(
        xh, Wkt, bk, Kh, (float*)nullptr, SLEN, HD, EMBED);
    gemm_bf16<<<(SLEN / 16) * (HD / 64) / 8, 256, 0, stream>>>(
        xh, Wvt, bv, Vh, (float*)nullptr, SLEN, HD, EMBED);

    // Fused flash MQA attention (async-LDS double-buffered K/V pipeline)
    mqa_flash<<<dim3(SLEN / 64, NHEADS), 128, 0, stream>>>(Qh, Kh, Vh, mask, Ah);

    // Output projection -> f32 result
    gemm_bf16<<<(SLEN / 16) * (EMBED / 64) / 8, 256, 0, stream>>>(
        Ah, Wot, bo, (unsigned short*)nullptr, out, SLEN, EMBED, EMBED);
}
